// FFTFeatureExtractor_82368882802941
// MI455X (gfx1250) — matrix operand order repaired and checked by reading, not compiled, not run on hardware
//
#include <hip/hip_runtime.h>
#include <math.h>

// ---------------------------------------------------------------------------
// FFT feature extractor for MI455X (gfx1250, wave32, WMMA + TDM)
//   fft2(1024x1024) -> log1p(|.|) -> per-sample minmax norm -> fftshift ->
//   bilinear resize to 224x224
// 1024-pt FFT = four-step 32x32: two 32x32 matmuls vs DFT-32 matrix; each
// 32x32 matmul = 2x2 tiles of V_WMMA_F32_16X16X32_F16 (K=32 in ONE wmma).
// DFT fragments + W1024 twiddles are wave-invariant: precomputed once into a
// 20KB L2-resident table (init kernel) and loaded per wave (b128/b64 hits)
// instead of ~1500 VALU sincos ops per wave.
// Column pass stages its 4KB-strided column via TENSOR_LOAD_TO_LDS (TDM),
// with D# LDS padding (1 dword per 32) giving conflict-free stride-33 rows.
// Workspace: re[48Mi] f32 | im[48Mi] f32 | mm[32] u32 | ftab 12KB | twd 8KB
// ---------------------------------------------------------------------------

typedef __attribute__((ext_vector_type(16))) _Float16 v16h;
typedef __attribute__((ext_vector_type(8)))  float    v8f;
typedef __attribute__((ext_vector_type(4)))  unsigned int u32x4;
typedef __attribute__((ext_vector_type(8)))  int      i32x8;
typedef __attribute__((ext_vector_type(4)))  int      i32x4;

#define NPLANES     48
#define PLANE_ELEMS (1024 * 1024)
#define OUT_HW      224
#define TWO_PI      6.2831853071795864f
#define TILE_F      1056                 /* 32 rows x 33 (padded) floats */

// fragment-table indices: B-operand DFT frags (cos/sin/-sin for nt=0,1),
// then A-operand DFT frags (cos/sin/-sin for mt=0,1)
#define FT_BC 0
#define FT_BS 2
#define FT_BSN 4
#define FT_AC 6
#define FT_AS 8
#define FT_ASN 10
#define FT_COUNT 12
#define FT_HALVES (FT_COUNT * 32 * 16)   /* 6144 halves = 12288 B */
#define TW_FLOATS (4 * 8 * 32 * 2)       /* 2048 floats = 8192 B  */

__device__ __forceinline__ int laneId() { return threadIdx.x & 31; }

// A-operand (16x32 f16) element mapping: lane L, packed half h -> (M, K)
__device__ __forceinline__ void a_coords(int L, int h, int* m, int* k) {
    *m = L & 15;
    *k = ((h >> 3) << 4) | (((L >> 4) & 1) << 3) | (h & 7);
}
// B-operand (32x16 f16) element mapping: lane L, packed half h -> (K, N)
__device__ __forceinline__ void b_coords(int L, int h, int* k, int* n) {
    *n = L & 15;
    *k = ((h >> 3) << 4) | (((L >> 4) & 1) << 3) | (h & 7);
}

__device__ __forceinline__ v8f wmma_f16(v16h a, v16h b, v8f c) {
    // D = A(16x32 f16) * B(32x16 f16) + C(f32)
    return __builtin_amdgcn_wmma_f32_16x16x32_f16(false, a, false, b,
                                                  (short)0, c, false, false);
}

// Load one precomputed per-lane fragment (16 halves = 32 B, coalesced b128s)
__device__ __forceinline__ v16h ld_frag(const _Float16* __restrict__ ftab, int idx) {
    return *(const v16h*)(ftab + (size_t)(idx * 32 + laneId()) * 16);
}

// Fragment loads from stride-33 LDS tiles (f32 -> f16 convert)
__device__ __forceinline__ v16h load_afrag_p(const float* x, int mt) {
    int L = laneId(); v16h a;                     // element (M,K) at x[M*33+K]
#pragma unroll
    for (int h = 0; h < 16; ++h) {
        int m, k; a_coords(L, h, &m, &k);
        a[h] = (_Float16)x[(m + 16 * mt) * 33 + k];
    }
    return a;
}
__device__ __forceinline__ v16h load_afrag_pT(const float* x, int mt) {
    int L = laneId(); v16h a;                     // element (M,K) at x[K*33+M]
#pragma unroll
    for (int h = 0; h < 16; ++h) {
        int m, k; a_coords(L, h, &m, &k);
        a[h] = (_Float16)x[k * 33 + (m + 16 * mt)];
    }
    return a;
}
__device__ __forceinline__ v16h load_bfrag_p(const float* x, int nt) {
    int L = laneId(); v16h b;                     // element (K,N) at x[K*33+N]
#pragma unroll
    for (int h = 0; h < 16; ++h) {
        int k, n; b_coords(L, h, &k, &n);
        b[h] = (_Float16)x[k * 33 + (n + 16 * nt)];
    }
    return b;
}

// ---------------------------------------------------------------------------
// TDM: DMA one 4KB-strided f32 column (1024 x 1 tile) into LDS, inserting
// 1 dword of padding every 32 dwords (=> stride-33 rows, transposed layout).
// D# built per CDNA5 ISA 8.3/8.4; groups 2/3 NULL-equivalent (2D tensor).
// ---------------------------------------------------------------------------
__device__ __forceinline__ void tdm_load_col(const float* gsrc, unsigned lds_byte_off) {
    unsigned long long ga = (unsigned long long)(size_t)gsrc;
    u32x4 g0;
    g0[0] = 1u;                                        // count=1, user D#
    g0[1] = lds_byte_off;                              // lds_addr (bytes)
    g0[2] = (unsigned)(ga & 0xFFFFFFFFu);              // global_addr[31:0]
    g0[3] = (unsigned)((ga >> 32) & 0x1FFFFFFu) | (2u << 30);  // [56:32] | type=2
    i32x8 g1;
    g1[0] = (2 << 16) | (1 << 20) | (4 << 22);         // data_size=4B, pad_en,
                                                       // pad_interval=32dw, pad=1dw
    g1[1] = (1024 << 16);                              // tensor_dim0[15:0]=1024
    g1[2] = (1024 << 16);                              // dim0 hi=0 | tensor_dim1 lo=1024
    g1[3] = (1 << 16);                                 // dim1 hi=0 | tile_dim0=1
    g1[4] = 1024;                                      // tile_dim1=1024 | tile_dim2=0
    g1[5] = 1024;                                      // tensor_dim0_stride[31:0]=1024
    g1[6] = 0;                                         // stride hi | dim1_stride lo
    g1[7] = 0;
    i32x4 zz = {0, 0, 0, 0};
#if defined(__clang_major__) && (__clang_major__ >= 23)
    i32x8 z8 = {0, 0, 0, 0, 0, 0, 0, 0};
    __builtin_amdgcn_tensor_load_to_lds(g0, g1, zz, zz, z8, 0);
#else
    __builtin_amdgcn_tensor_load_to_lds(g0, g1, zz, zz, 0);
#endif
}

// ---------------------------------------------------------------------------
// Init kernel: min/max seeds + constant fragment/twiddle tables (1 wave).
// ---------------------------------------------------------------------------
__global__ void init_kernel(unsigned int* __restrict__ mm,
                            _Float16* __restrict__ ftab,
                            float* __restrict__ twd) {
    const int L = threadIdx.x;                          // 0..31
    if (L < 16) { mm[L] = 0x7F800000u; mm[16 + L] = 0u; }
    // B-operand DFT-32 frags: element (K,N), Fr=cos, Fi=-sin
    for (int nt = 0; nt < 2; ++nt)
        for (int h = 0; h < 16; ++h) {
            int k, n; b_coords(L, h, &k, &n);
            float th = (float)((k * (n + 16 * nt)) & 31) * (TWO_PI / 32.f);
            float s, c; __sincosf(th, &s, &c);
            ftab[(size_t)((FT_BC  + nt) * 32 + L) * 16 + h] = (_Float16)c;
            ftab[(size_t)((FT_BS  + nt) * 32 + L) * 16 + h] = (_Float16)s;
            ftab[(size_t)((FT_BSN + nt) * 32 + L) * 16 + h] = (_Float16)(-s);
        }
    // A-operand DFT-32 frags: element (M,K)
    for (int mt = 0; mt < 2; ++mt)
        for (int h = 0; h < 16; ++h) {
            int m, k; a_coords(L, h, &m, &k);
            float th = (float)(((m + 16 * mt) * k) & 31) * (TWO_PI / 32.f);
            float s, c; __sincosf(th, &s, &c);
            ftab[(size_t)((FT_AC  + mt) * 32 + L) * 16 + h] = (_Float16)c;
            ftab[(size_t)((FT_AS  + mt) * 32 + L) * 16 + h] = (_Float16)s;
            ftab[(size_t)((FT_ASN + mt) * 32 + L) * 16 + h] = (_Float16)(-s);
        }
    // W1024 twiddles at the C/D-fragment positions: [mt][nt][r][lane]{cos,sin}
    for (int mt = 0; mt < 2; ++mt)
        for (int nt = 0; nt < 2; ++nt)
            for (int r = 0; r < 8; ++r) {
                int n1 = 16 * mt + r + 8 * (L >> 4);
                int k1 = 16 * nt + (L & 15);
                float ph = (float)((n1 * k1) & 1023) * (TWO_PI / 1024.f);
                float s, c; __sincosf(ph, &s, &c);
                int o = (((mt * 2 + nt) * 8 + r) * 32 + L) * 2;
                twd[o] = c; twd[o + 1] = s;
            }
}

// ---------------------------------------------------------------------------
// Kernel 1: 1024-pt FFT of every row (real input). One wave per row.
// x[n1][n2] = row[n1 + 32*n2];  G = X*F;  H = G .* W1024^(n1*k1);  Y = F*H;
// out[k1 + 32*k2] = Y[k2][k1]
// ---------------------------------------------------------------------------
__global__ __launch_bounds__(128) void rowfft_kernel(const float* __restrict__ img,
                                                     float* __restrict__ re,
                                                     float* __restrict__ im,
                                                     const _Float16* __restrict__ ftab,
                                                     const float* __restrict__ twd) {
    __shared__ float sx[4][TILE_F];
    __shared__ float shr[4][TILE_F];
    __shared__ float shi[4][TILE_F];
    const int w = threadIdx.x >> 5, L = threadIdx.x & 31;
    const long row = (long)blockIdx.x * 4 + w;          // 0 .. 48*1024-1
    const float* src = img + row * 1024;
    const float2* tw2 = (const float2*)twd;

#pragma unroll 4
    for (int t = 0; t < 32; ++t) {
        sx[w][L * 33 + t] = src[t * 32 + L];            // coalesced
        __builtin_prefetch(src + t * 32 + L + 1024, 0, 0);  // global_prefetch_b8
    }

    const v8f z = {0.f, 0.f, 0.f, 0.f, 0.f, 0.f, 0.f, 0.f};

    // Stage 1: G = X * F (X real).  Fr = cos, Fi = -sin.
    v16h bc[2], bsn[2];
#pragma unroll
    for (int nt = 0; nt < 2; ++nt) {
        bc[nt]  = ld_frag(ftab, FT_BC + nt);
        bsn[nt] = ld_frag(ftab, FT_BSN + nt);
    }

#pragma unroll
    for (int mt = 0; mt < 2; ++mt) {
        v16h ax = load_afrag_p(sx[w], mt);
#pragma unroll
        for (int nt = 0; nt < 2; ++nt) {
            v8f gr = wmma_f16(ax, bc[nt],  z);          // X * cos
            v8f gi = wmma_f16(ax, bsn[nt], z);          // X * (-sin)
#pragma unroll
            for (int r = 0; r < 8; ++r) {               // twiddle, store H
                int n1 = 16 * mt + r + 8 * (L >> 4);
                int k1 = 16 * nt + (L & 15);
                float2 t = tw2[((mt * 2 + nt) * 8 + r) * 32 + L];
                shr[w][n1 * 33 + k1] = gr[r] * t.x + gi[r] * t.y;
                shi[w][n1 * 33 + k1] = gi[r] * t.x - gr[r] * t.y;
            }
        }
    }

    // Stage 2: Y = F * H.  Yr = cos*Hr + sin*Hi ; Yi = cos*Hi - sin*Hr
    float* reDst = re + row * 1024;
    float* imDst = im + row * 1024;
#pragma unroll
    for (int mt = 0; mt < 2; ++mt) {
        v16h fc  = ld_frag(ftab, FT_AC + mt);
        v16h fs  = ld_frag(ftab, FT_AS + mt);
        v16h fsn = ld_frag(ftab, FT_ASN + mt);
#pragma unroll
        for (int nt = 0; nt < 2; ++nt) {
            v16h bhr = load_bfrag_p(shr[w], nt);
            v16h bhi = load_bfrag_p(shi[w], nt);
            v8f yr = wmma_f16(fs,  bhi, wmma_f16(fc, bhr, z));
            v8f yi = wmma_f16(fsn, bhr, wmma_f16(fc, bhi, z));
#pragma unroll
            for (int r = 0; r < 8; ++r) {
                int k2 = 16 * mt + r + 8 * (L >> 4);
                int k1 = 16 * nt + (L & 15);
                int idx = k1 + 32 * k2;
                reDst[idx] = yr[r];
                imDst[idx] = yi[r];
            }
        }
    }
}

// ---------------------------------------------------------------------------
// Kernel 2: 1024-pt FFT of every column (complex), column staged via TDM.
// Fused with spec = log1p(|Y|) (in-place over `re`) and per-sample min/max.
// Dynamic LDS: 4 waves x (re tile + im tile) x 1056 floats = 33792 B.
// ---------------------------------------------------------------------------
__global__ __launch_bounds__(128) void colfft_kernel(float* __restrict__ re,
                                                     float* __restrict__ im,
                                                     unsigned int* __restrict__ mm,
                                                     const _Float16* __restrict__ ftab,
                                                     const float* __restrict__ twd) {
    extern __shared__ float smem[];
    const int w = threadIdx.x >> 5, L = threadIdx.x & 31;
    float* tr = smem + w * (2 * TILE_F);                // xT[n2][n1], stride 33
    float* ti = tr + TILE_F;
    const long colg = (long)blockIdx.x * 4 + w;
    const int p = (int)(colg >> 10);                    // plane 0..47
    const int c = (int)(colg & 1023);                   // column
    float* base_r = re + (long)p * PLANE_ELEMS;
    float* base_i = im + (long)p * PLANE_ELEMS;
    const float2* tw2 = (const float2*)twd;

    // Async DMA of the strided column into LDS (TENSORcnt-tracked).
    unsigned lds0 = __builtin_amdgcn_groupstaticsize() +
                    (unsigned)(w * (2 * TILE_F) * 4);
    tdm_load_col(base_r + c, lds0);
    tdm_load_col(base_i + c, lds0 + TILE_F * 4);
    __builtin_amdgcn_s_wait_tensorcnt(0);
    asm volatile("" ::: "memory");

    const v8f z = {0.f, 0.f, 0.f, 0.f, 0.f, 0.f, 0.f, 0.f};

    // Consume all LDS data into A-fragments before reusing tr/ti for H.
    // TDM wrote column element r=n1+32*n2 linearly => transposed tile.
    v16h ar[2], ai[2];
#pragma unroll
    for (int mt = 0; mt < 2; ++mt) {
        ar[mt] = load_afrag_pT(tr, mt);
        ai[mt] = load_afrag_pT(ti, mt);
    }
    v16h bc[2], bs[2], bsn[2];
#pragma unroll
    for (int nt = 0; nt < 2; ++nt) {
        bc[nt]  = ld_frag(ftab, FT_BC + nt);
        bs[nt]  = ld_frag(ftab, FT_BS + nt);
        bsn[nt] = ld_frag(ftab, FT_BSN + nt);
    }

    // Stage 1 (complex): Gr = Ar*cos + Ai*sin ; Gi = Ai*cos - Ar*sin
#pragma unroll
    for (int mt = 0; mt < 2; ++mt)
#pragma unroll
        for (int nt = 0; nt < 2; ++nt) {
            v8f gr = wmma_f16(ai[mt], bs[nt],  wmma_f16(ar[mt], bc[nt], z));
            v8f gi = wmma_f16(ar[mt], bsn[nt], wmma_f16(ai[mt], bc[nt], z));
#pragma unroll
            for (int r = 0; r < 8; ++r) {               // twiddle, store H
                int n1 = 16 * mt + r + 8 * (L >> 4);
                int k1 = 16 * nt + (L & 15);
                float2 t = tw2[((mt * 2 + nt) * 8 + r) * 32 + L];
                tr[n1 * 33 + k1] = gr[r] * t.x + gi[r] * t.y;
                ti[n1 * 33 + k1] = gi[r] * t.x - gr[r] * t.y;
            }
        }

    // Stage 2 + spectrum + min/max
    float lmin = __uint_as_float(0x7F800000u), lmax = 0.f;   // spec >= 0
#pragma unroll
    for (int mt = 0; mt < 2; ++mt) {
        v16h fc  = ld_frag(ftab, FT_AC + mt);
        v16h fs  = ld_frag(ftab, FT_AS + mt);
        v16h fsn = ld_frag(ftab, FT_ASN + mt);
#pragma unroll
        for (int nt = 0; nt < 2; ++nt) {
            v16h bhr = load_bfrag_p(tr, nt);
            v16h bhi = load_bfrag_p(ti, nt);
            v8f yr = wmma_f16(fs,  bhi, wmma_f16(fc, bhr, z));
            v8f yi = wmma_f16(fsn, bhr, wmma_f16(fc, bhi, z));
#pragma unroll
            for (int r = 0; r < 8; ++r) {
                int k2 = 16 * mt + r + 8 * (L >> 4);
                int k1 = 16 * nt + (L & 15);
                int kk = k1 + 32 * k2;
                float sp = log1pf(sqrtf(yr[r] * yr[r] + yi[r] * yi[r]));
                base_r[(long)kk * 1024 + c] = sp;       // spec aliases re plane
                lmin = fminf(lmin, sp);
                lmax = fmaxf(lmax, sp);
            }
        }
    }

    // wave32 butterfly reduce, then uint-bit atomics (spec >= 0 => ordered)
#pragma unroll
    for (int off = 16; off > 0; off >>= 1) {
        lmin = fminf(lmin, __shfl_xor(lmin, off, 32));
        lmax = fmaxf(lmax, __shfl_xor(lmax, off, 32));
    }
    if (L == 0) {
        int b = p / 3;
        atomicMin(&mm[b],      __float_as_uint(lmin));
        atomicMax(&mm[16 + b], __float_as_uint(lmax));
    }
}

// ---------------------------------------------------------------------------
// Kernel 3: fftshift (folded into gather) + normalize + bilinear 1024 -> 224
__global__ __launch_bounds__(256) void resize_kernel(const float* __restrict__ spec,
                                                     const unsigned int* __restrict__ mm,
                                                     float* __restrict__ out, int total) {
    int idx = blockIdx.x * 256 + threadIdx.x;
    if (idx >= total) return;
    int ox = idx % OUT_HW;
    int oy = (idx / OUT_HW) % OUT_HW;
    int pl = idx / (OUT_HW * OUT_HW);                   // plane 0..47
    int b = pl / 3;
    float mn = __uint_as_float(mm[b]);
    float mx = __uint_as_float(mm[16 + b]);
    float inv = 1.f / (mx - mn + 1e-8f);
    const float scale = 1024.f / 224.f;                 // half-pixel centers
    float sy = ((float)oy + 0.5f) * scale - 0.5f;
    float sx = ((float)ox + 0.5f) * scale - 0.5f;
    int y0 = (int)floorf(sy), x0 = (int)floorf(sx);
    float wy = sy - (float)y0, wx = sx - (float)x0;
    int y1 = y0 + 1, x1 = x0 + 1;
    y0 = y0 < 0 ? 0 : (y0 > 1023 ? 1023 : y0);
    y1 = y1 < 0 ? 0 : (y1 > 1023 ? 1023 : y1);
    x0 = x0 < 0 ? 0 : (x0 > 1023 ? 1023 : x0);
    x1 = x1 < 0 ? 0 : (x1 > 1023 ? 1023 : x1);
    int Y0 = (y0 + 512) & 1023, Y1 = (y1 + 512) & 1023; // fftshift fold
    int X0 = (x0 + 512) & 1023, X1 = (x1 + 512) & 1023;
    const float* pb = spec + (long)pl * PLANE_ELEMS;
    float v00 = pb[(long)Y0 * 1024 + X0];
    float v01 = pb[(long)Y0 * 1024 + X1];
    float v10 = pb[(long)Y1 * 1024 + X0];
    float v11 = pb[(long)Y1 * 1024 + X1];
    float v0 = v00 + (v01 - v00) * wx;
    float v1 = v10 + (v11 - v10) * wx;
    float v = v0 + (v1 - v0) * wy;
    out[idx] = (v - mn) * inv;
}

// ---------------------------------------------------------------------------
extern "C" void kernel_launch(void* const* d_in, const int* in_sizes, int n_in,
                              void* d_out, int out_size, void* d_ws, size_t ws_size,
                              hipStream_t stream) {
    const float* images = (const float*)d_in[0];        // (16,3,1024,1024) f32
    float* out = (float*)d_out;                         // (16,3,224,224) f32
    float* re = (float*)d_ws;                           // 48 Mi floats
    float* im = re + (size_t)NPLANES * PLANE_ELEMS;     // 48 Mi floats
    unsigned int* mm = (unsigned int*)(im + (size_t)NPLANES * PLANE_ELEMS);
    _Float16* ftab = (_Float16*)(mm + 32);              // 12288 B, 32B-aligned
    float* twd = (float*)(ftab + FT_HALVES);            // 8192 B

    init_kernel<<<1, 32, 0, stream>>>(mm, ftab, twd);
    rowfft_kernel<<<(NPLANES * 1024) / 4, 128, 0, stream>>>(images, re, im, ftab, twd);
    colfft_kernel<<<(NPLANES * 1024) / 4, 128,
                    4 * (2 * TILE_F) * sizeof(float), stream>>>(re, im, mm, ftab, twd);
    resize_kernel<<<(out_size + 255) / 256, 256, 0, stream>>>(re, mm, out, out_size);
}
